// MMRecBlock_9491877724593
// MI455X (gfx1250) — matrix operand order repaired and checked by reading, not compiled, not run on hardware
//
#include <hip/hip_runtime.h>
#include <hip/hip_bf16.h>
#include <math.h>

// Problem constants (match reference: B,S,D,H,FF,R = 4,1024,1024,16,4096,128)
constexpr int NB  = 4;
constexpr int NS  = 1024;
constexpr int ND  = 1024;
constexpr int NH  = 16;
constexpr int NFF = 4096;
constexpr int NR  = 128;
constexpr int NDH = 64;
constexpr int NM  = NB * NS;   // 4096 rows for all GEMMs

typedef __bf16 v16bf __attribute__((ext_vector_type(16)));
typedef float  v8f   __attribute__((ext_vector_type(8)));
typedef int    v4i   __attribute__((vector_size(16)));   // matches builtin param type

union FragBF {
  v16bf v;
  float4 f[2];
};

__device__ __forceinline__ v8f wmma_bf16(const FragBF& a, const FragBF& b, v8f c) {
  // D = A(16x32 bf16) * B(32x16 bf16) + C(16x16 f32)
  return __builtin_amdgcn_wmma_f32_16x16x32_bf16(false, a.v, false, b.v, (short)0, c,
                                                 false, false);
}

// ---------------------------------------------------------------------------
// CDNA5 async global->LDS copy (16 bytes per lane) + ASYNCcnt wait.
// Builtin signature (from clang diagnostic): param0 = v4i __device__(AS1)*,
// so pass exactly (AS1 v4i*, AS3 v4i*, imm offset, imm cpol).
// LDS AS(3) address = low 32 bits of the flat address (ISA 10.2 aperture rule).
// ---------------------------------------------------------------------------
__device__ __forceinline__ void async_copy_b128(const void* g, void* l) {
#if __has_builtin(__builtin_amdgcn_global_load_async_to_lds_b128)
  __builtin_amdgcn_global_load_async_to_lds_b128(
      (__attribute__((address_space(1))) v4i*)(uintptr_t)g,
      (__attribute__((address_space(3))) v4i*)(uint32_t)(uintptr_t)l,
      0, 0);
#else
  asm volatile("global_load_async_to_lds_b128 %0, %1, off"
               :
               : "v"((uint32_t)(uintptr_t)l), "v"((uint64_t)(uintptr_t)g)
               : "memory");
#endif
}

__device__ __forceinline__ void wait_asynccnt0() {
#if __has_builtin(__builtin_amdgcn_s_wait_asynccnt)
  __builtin_amdgcn_s_wait_asynccnt(0);
#else
  asm volatile("s_wait_asynccnt 0" ::: "memory");
#endif
}

// ---------------------------------------------------------------------------
// f32 -> bf16 convert (weights etc.)
// ---------------------------------------------------------------------------
__global__ void cvt_bf16_kernel(const float* __restrict__ in,
                                __hip_bfloat16* __restrict__ out, size_t n) {
  size_t i = (size_t)blockIdx.x * blockDim.x + threadIdx.x;
  size_t stride = (size_t)gridDim.x * blockDim.x;
  for (; i < n; i += stride) out[i] = __float2bfloat16(in[i]);
}

// ---------------------------------------------------------------------------
// RMSNorm (per row of length ND), bf16 output
// norm = ||x||/sqrt(D); out = w * x / (norm + eps)
// ---------------------------------------------------------------------------
__global__ __launch_bounds__(256)
void rmsnorm_bf16_kernel(const float* __restrict__ x, const float* __restrict__ w,
                         __hip_bfloat16* __restrict__ out) {
  const int row = blockIdx.x;
  const int tid = threadIdx.x;
  const size_t base = (size_t)row * ND;
  __shared__ float red[256];
  float acc = 0.f;
  for (int i = tid; i < ND; i += 256) { float v = x[base + i]; acc += v * v; }
  red[tid] = acc;
  __syncthreads();
  for (int s = 128; s > 0; s >>= 1) {
    if (tid < s) red[tid] += red[tid + s];
    __syncthreads();
  }
  const float norm = sqrtf(red[0]) * (1.0f / 32.0f);  // sqrt(D)=32
  const float inv = 1.0f / (norm + 1e-6f);
  for (int i = tid; i < ND; i += 256)
    out[base + i] = __float2bfloat16(w[i] * x[base + i] * inv);
}

// ---------------------------------------------------------------------------
// WMMA bf16 GEMM with double-buffered async global->LDS staging.
//   C[M,N] = act(A[M,K] @ W[N,K]^T + bias) * scale (+ res)
// op: 0=none, 1=relu, 2=gelu(tanh), 3=sigmoid+clamp
// Block: 256 threads = 8 waves; wave tile 32x32 (2x2 WMMA tiles); block 64x128.
// Per 32-K step: A-tile 64x32 (4KB) + B-tile 128x32 (8KB) staged via
// GLOBAL_LOAD_ASYNC_TO_LDS_B128, consumed via ds_load_b128 fragments.
// ---------------------------------------------------------------------------
__device__ __forceinline__ void gemm_store(v8f c, int mBase, int nBase, int lh, int l15,
                                           const float* __restrict__ bias,
                                           float* __restrict__ outF,
                                           __hip_bfloat16* __restrict__ outB,
                                           const float* __restrict__ res,
                                           int N, int op, float scale) {
  const int n = nBase + l15;
  const float bv = bias ? bias[n] : 0.f;
#pragma unroll
  for (int v = 0; v < 8; ++v) {
    const int m = mBase + v + 8 * lh;
    float x = c[v] + bv;
    if (op == 1) {
      x = fmaxf(x, 0.f);
    } else if (op == 2) {
      float u = 0.7978845608028654f * (x + 0.044715f * x * x * x);
      x = 0.5f * x * (1.0f + tanhf(u));
    } else if (op == 3) {
      x = 1.0f / (1.0f + __expf(-x));
      x = fminf(fmaxf(x, 1e-6f), 1.0f - 1e-6f);
    }
    x *= scale;
    const size_t idx = (size_t)m * N + n;
    if (res) x += res[idx];
    if (outF) outF[idx] = x;
    if (outB) outB[idx] = __float2bfloat16(x);
  }
}

__global__ __launch_bounds__(256)
void gemm_bf16_kernel(const __hip_bfloat16* __restrict__ A,
                      const __hip_bfloat16* __restrict__ W,
                      const float* __restrict__ bias,
                      float* __restrict__ outF,
                      __hip_bfloat16* __restrict__ outB,
                      const float* __restrict__ res,
                      int M, int N, int K, int op, float scale) {
  const int tid = threadIdx.x;
  const int wave = tid >> 5;
  const int lane = tid & 31;
  const int lh = lane >> 4;
  const int l15 = lane & 15;
  const int wm = wave >> 2;               // 0..1
  const int wn = wave & 3;                // 0..3
  const int rowBlk = blockIdx.y * 64;
  const int colBlk = blockIdx.x * 128;
  const int kbA = lh ? 8 : 0;             // A-fragment K interleave (sec 7.12.2)
  const int kbB = lh * 8;                 // B-fragment K interleave (matches A)

  // Double-buffered staging tiles: A 64x32, B 128x32 (bf16)
  __shared__ __align__(16) __hip_bfloat16 sA[2][64 * 32];   // 2 x 4KB
  __shared__ __align__(16) __hip_bfloat16 sB[2][128 * 32];  // 2 x 8KB

  // Per-thread staging chunk coordinates (16B = 8 bf16 per async op)
  const int ar = tid >> 2;                 // A row 0..63
  const int ac = (tid & 3) * 8;            // A k-chunk
  const int br0 = tid >> 2;                // B rows 0..63 (chunk set 0)
  const int br1 = (tid + 256) >> 2;        // B rows 64..127 (chunk set 1)

  const __hip_bfloat16* gA = A + (size_t)(rowBlk + ar) * K + ac;
  const __hip_bfloat16* gB0 = W + (size_t)(colBlk + br0) * K + ac;
  const __hip_bfloat16* gB1 = W + (size_t)(colBlk + br1) * K + ac;

  auto stage = [&](int k0, int buf) {
    async_copy_b128(gA + k0, &sA[buf][ar * 32 + ac]);
    async_copy_b128(gB0 + k0, &sB[buf][br0 * 32 + ac]);
    async_copy_b128(gB1 + k0, &sB[buf][br1 * 32 + ac]);
  };

  stage(0, 0);

  v8f c00 = {}, c01 = {}, c10 = {}, c11 = {};
  int buf = 0;
  for (int k0 = 0; k0 < K; k0 += 32) {
    wait_asynccnt0();        // this wave's stage landed in LDS
    __syncthreads();         // all waves staged; prior ds reads fenced
    if (k0 + 32 < K) stage(k0 + 32, buf ^ 1);

    FragBF a0, a1, b0, b1;
    const __hip_bfloat16* pa = &sA[buf][(wm * 32 + l15) * 32 + kbA];
    a0.f[0] = *(const float4*)pa;
    a0.f[1] = *(const float4*)(pa + 16);
    a1.f[0] = *(const float4*)(pa + 16 * 32);
    a1.f[1] = *(const float4*)(pa + 16 * 32 + 16);
    const __hip_bfloat16* pb = &sB[buf][(wn * 32 + l15) * 32 + kbB];
    b0.f[0] = *(const float4*)pb;
    b0.f[1] = *(const float4*)(pb + 16);
    b1.f[0] = *(const float4*)(pb + 16 * 32);
    b1.f[1] = *(const float4*)(pb + 16 * 32 + 16);

    c00 = wmma_bf16(a0, b0, c00);
    c01 = wmma_bf16(a0, b1, c01);
    c10 = wmma_bf16(a1, b0, c10);
    c11 = wmma_bf16(a1, b1, c11);
    buf ^= 1;
  }

  const int row0 = rowBlk + wm * 32;
  const int col0 = colBlk + wn * 32;
  gemm_store(c00, row0,      col0,      lh, l15, bias, outF, outB, res, N, op, scale);
  gemm_store(c01, row0,      col0 + 16, lh, l15, bias, outF, outB, res, N, op, scale);
  gemm_store(c10, row0 + 16, col0,      lh, l15, bias, outF, outB, res, N, op, scale);
  gemm_store(c11, row0 + 16, col0 + 16, lh, l15, bias, outF, outB, res, N, op, scale);
}

// ---------------------------------------------------------------------------
// Transpose V: v_bf[b,s,h*DH+dh] -> vt[(b*H+h)*DH+dh, s]   (bf16)
// Makes the PV B-operand contiguous along K(=kv).
// ---------------------------------------------------------------------------
__global__ void transpose_v_kernel(const __hip_bfloat16* __restrict__ vin,
                                   __hip_bfloat16* __restrict__ vt) {
  const size_t n = (size_t)NB * NS * ND;
  size_t i = (size_t)blockIdx.x * blockDim.x + threadIdx.x;
  size_t stride = (size_t)gridDim.x * blockDim.x;
  for (; i < n; i += stride) {
    size_t b = i / ((size_t)NS * ND);
    size_t rem = i % ((size_t)NS * ND);
    size_t s = rem / ND;
    size_t c = rem % ND;
    size_t h = c / NDH;
    size_t dh = c % NDH;
    vt[(((size_t)b * NH + h) * NDH + dh) * NS + s] = vin[i];
  }
}

// ---------------------------------------------------------------------------
// Sequential recurrence: h_t = z_t * sigmoid(h_{t-1} @ Wg^T + bg) + gamma_t * h_{t-1}
// One block per batch; hidden state lives in LDS; Wg streamed from L2 (4 MB).
// ---------------------------------------------------------------------------
__global__ __launch_bounds__(1024)
void recurrence_kernel(const float* __restrict__ z, const float* __restrict__ gmm,
                       const float* __restrict__ Wg, const float* __restrict__ bg,
                       float* __restrict__ hseq) {
  const int b = blockIdx.x;
  const int tid = threadIdx.x;
  __shared__ float h[ND];
  h[tid] = 0.f;
  __syncthreads();
  const float* wrow = Wg + (size_t)tid * ND;
  for (int t = 0; t < NS; ++t) {
    float acc = bg[tid];
#pragma unroll 4
    for (int k = 0; k < ND; k += 4) {
      float4 hv = *(const float4*)&h[k];
      float4 wv = *(const float4*)&wrow[k];
      acc += hv.x * wv.x + hv.y * wv.y + hv.z * wv.z + hv.w * wv.w;
    }
    const float sig = 1.0f / (1.0f + __expf(-acc));
    const size_t idx = ((size_t)b * NS + t) * ND + tid;
    const float hn = z[idx] * sig + gmm[idx] * h[tid];
    __syncthreads();
    h[tid] = hn;
    hseq[idx] = hn;
    __syncthreads();
  }
}

// ---------------------------------------------------------------------------
// Attention: one block per (q-tile of 16, head, batch).
// Pass1: WMMA QK^T (Q pre-scaled by 1/8) -> bf16 score strip in LDS.
// Softmax over the 1024-wide strip in LDS.
// Pass2: WMMA P @ V (V transposed) -> bf16 output (heads interleaved layout).
// ---------------------------------------------------------------------------
__global__ __launch_bounds__(256)
void attention_kernel(const __hip_bfloat16* __restrict__ q,
                      const __hip_bfloat16* __restrict__ k,
                      const __hip_bfloat16* __restrict__ vt,
                      __hip_bfloat16* __restrict__ outB) {
  const int qt = blockIdx.x;   // 0..S/16-1
  const int h  = blockIdx.y;   // 0..NH-1
  const int b  = blockIdx.z;   // 0..NB-1
  const int tid = threadIdx.x;
  const int wave = tid >> 5;
  const int lane = tid & 31;
  const int lh = lane >> 4;
  const int l15 = lane & 15;
  const int kbA = lh ? 8 : 0;

  __shared__ __align__(32) __hip_bfloat16 sP[16 * NS];  // 32 KB score/prob strip
  __shared__ float rpart[16 * 16];
  __shared__ float rowmax[16];
  __shared__ float rowsum[16];

  const int s0 = qt * 16;

  // Q fragments for this 16-row tile: K(=dh) = 0..31 and 32..63
  const size_t qbase = ((size_t)(b * NS + s0 + l15)) * ND + h * NDH;
  FragBF aq0, aq1;
  aq0.f[0] = *(const float4*)(q + qbase + kbA);
  aq0.f[1] = *(const float4*)(q + qbase + kbA + 16);
  aq1.f[0] = *(const float4*)(q + qbase + 32 + kbA);
  aq1.f[1] = *(const float4*)(q + qbase + 32 + kbA + 16);

  // ---- Pass 1: scores = (Q/8) K^T, written bf16 into LDS strip ----
  for (int kt = wave; kt < NS / 16; kt += 8) {
    const int kv0 = kt * 16;
    const size_t kbase = ((size_t)(b * NS + kv0 + l15)) * ND + h * NDH + lh * 8;
    FragBF b0, b1;
    b0.f[0] = *(const float4*)(k + kbase);
    b0.f[1] = *(const float4*)(k + kbase + 16);
    b1.f[0] = *(const float4*)(k + kbase + 32);
    b1.f[1] = *(const float4*)(k + kbase + 48);
    v8f c = {};
    c = wmma_bf16(aq0, b0, c);
    c = wmma_bf16(aq1, b1, c);
#pragma unroll
    for (int v = 0; v < 8; ++v) {
      const int m = v + 8 * lh;
      sP[m * NS + kv0 + l15] = __float2bfloat16(c[v]);
    }
  }
  __syncthreads();

  // ---- Softmax over each 1024-wide row of the strip ----
  const int r = tid >> 4;    // row 0..15
  const int sub = tid & 15;  // 16 threads per row
  float mx = -1e30f;
  for (int j = sub; j < NS; j += 16)
    mx = fmaxf(mx, __bfloat162float(sP[r * NS + j]));
  rpart[r * 16 + sub] = mx;
  __syncthreads();
  if (sub == 0) {
    float m2 = rpart[r * 16];
    for (int j = 1; j < 16; ++j) m2 = fmaxf(m2, rpart[r * 16 + j]);
    rowmax[r] = m2;
  }
  __syncthreads();
  const float mrow = rowmax[r];
  float sum = 0.f;
  for (int j = sub; j < NS; j += 16) {
    const float e = __expf(__bfloat162float(sP[r * NS + j]) - mrow);
    sum += e;
    sP[r * NS + j] = __float2bfloat16(e);
  }
  rpart[r * 16 + sub] = sum;
  __syncthreads();
  if (sub == 0) {
    float t = 0.f;
    for (int j = 0; j < 16; ++j) t += rpart[r * 16 + j];
    rowsum[r] = t;
  }
  __syncthreads();

  // ---- Pass 2: O = P @ V ; waves 0..3 each own a 16-wide dh slice ----
  if (wave < 4) {
    const int dh0 = wave * 16;
    v8f o = {};
    const size_t vrow = ((size_t)(b * NH + h) * NDH + dh0 + l15) * NS;
    for (int k0 = 0; k0 < NS; k0 += 32) {
      FragBF a, bb;
      const __hip_bfloat16* pa = &sP[l15 * NS + k0 + kbA];
      a.f[0] = *(const float4*)pa;
      a.f[1] = *(const float4*)(pa + 16);
      const __hip_bfloat16* pb = vt + vrow + k0 + lh * 8;
      bb.f[0] = *(const float4*)pb;
      bb.f[1] = *(const float4*)(pb + 16);
      o = wmma_bf16(a, bb, o);
    }
#pragma unroll
    for (int v = 0; v < 8; ++v) {
      const int m = v + 8 * lh;
      const float val = o[v] / rowsum[m];
      const size_t oi = ((size_t)(b * NS + s0 + m)) * ND + h * NDH + dh0 + l15;
      outB[oi] = __float2bfloat16(val);
    }
  }
}

// ---------------------------------------------------------------------------
// y = x + ao + hseq
// ---------------------------------------------------------------------------
__global__ void add3_kernel(const float* __restrict__ a, const float* __restrict__ b,
                            const float* __restrict__ c, float* __restrict__ out,
                            size_t n) {
  size_t i = (size_t)blockIdx.x * blockDim.x + threadIdx.x;
  size_t stride = (size_t)gridDim.x * blockDim.x;
  for (; i < n; i += stride) out[i] = a[i] + b[i] + c[i];
}

// ---------------------------------------------------------------------------
// Host-side orchestration
// ---------------------------------------------------------------------------
extern "C" void kernel_launch(void* const* d_in, const int* in_sizes, int n_in,
                              void* d_out, int out_size, void* d_ws, size_t ws_size,
                              hipStream_t stream) {
  (void)in_sizes; (void)n_in; (void)out_size; (void)ws_size;

  const float* x    = (const float*)d_in[0];
  const float* n1w  = (const float*)d_in[1];
  const float* Wq   = (const float*)d_in[2];
  const float* bq   = (const float*)d_in[3];
  const float* Wk   = (const float*)d_in[4];
  const float* bk   = (const float*)d_in[5];
  const float* Wv   = (const float*)d_in[6];
  const float* bv   = (const float*)d_in[7];
  const float* Wz   = (const float*)d_in[8];
  const float* bz   = (const float*)d_in[9];
  const float* Wg   = (const float*)d_in[10];
  const float* bg   = (const float*)d_in[11];
  const float* gdw  = (const float*)d_in[12];
  const float* gdb  = (const float*)d_in[13];
  const float* guw  = (const float*)d_in[14];
  const float* gub  = (const float*)d_in[15];
  const float* Wo   = (const float*)d_in[16];
  const float* bo   = (const float*)d_in[17];
  const float* n2w  = (const float*)d_in[18];
  const float* f1w  = (const float*)d_in[19];
  const float* f1b  = (const float*)d_in[20];
  const float* f2w  = (const float*)d_in[21];
  const float* f2b  = (const float*)d_in[22];
  float* out = (float*)d_out;

  char* ws = (char*)d_ws;
  size_t off = 0;
  auto alloc = [&](size_t bytes) -> char* {
    char* p = ws + off;
    off += (bytes + 255) & ~(size_t)255;
    return p;
  };

  const size_t MD2 = (size_t)NM * ND * 2;
  const size_t MD4 = (size_t)NM * ND * 4;

  __hip_bfloat16* xn_bf = (__hip_bfloat16*)alloc(MD2);  // reused as yn_bf
  __hip_bfloat16* q_bf  = (__hip_bfloat16*)alloc(MD2);  // q..vt reused as ff1 (32MB)
  __hip_bfloat16* k_bf  = (__hip_bfloat16*)alloc(MD2);
  __hip_bfloat16* v_bf  = (__hip_bfloat16*)alloc(MD2);
  __hip_bfloat16* vt_bf = (__hip_bfloat16*)alloc(MD2);
  float*          z_f   = (float*)alloc(MD4);           // reused as ao_f
  __hip_bfloat16* z_bf  = (__hip_bfloat16*)alloc(MD2);  // reused as aob_bf
  __hip_bfloat16* g1_bf = (__hip_bfloat16*)alloc((size_t)NM * NR * 2);
  float*          gamma = (float*)alloc(MD4);           // reused as y
  float*          hseq  = (float*)alloc(MD4);

  __hip_bfloat16* wq_bf  = (__hip_bfloat16*)alloc((size_t)ND * ND * 2);
  __hip_bfloat16* wk_bf  = (__hip_bfloat16*)alloc((size_t)ND * ND * 2);
  __hip_bfloat16* wv_bf  = (__hip_bfloat16*)alloc((size_t)ND * ND * 2);
  __hip_bfloat16* wz_bf  = (__hip_bfloat16*)alloc((size_t)ND * ND * 2);
  __hip_bfloat16* wo_bf  = (__hip_bfloat16*)alloc((size_t)ND * ND * 2);
  __hip_bfloat16* gdw_bf = (__hip_bfloat16*)alloc((size_t)NR * ND * 2);
  __hip_bfloat16* guw_bf = (__hip_bfloat16*)alloc((size_t)ND * NR * 2);
  __hip_bfloat16* f1w_bf = (__hip_bfloat16*)alloc((size_t)NFF * ND * 2);
  __hip_bfloat16* f2w_bf = (__hip_bfloat16*)alloc((size_t)ND * NFF * 2);

  __hip_bfloat16* yn_bf  = xn_bf;
  __hip_bfloat16* ff1_bf = q_bf;  // spans q/k/v/vt buffers (32MB contiguous)
  __hip_bfloat16* aob_bf = z_bf;
  float* ao_f = z_f;
  float* y    = gamma;

  auto cvt = [&](const float* s, __hip_bfloat16* d, size_t n) {
    cvt_bf16_kernel<<<dim3(1024), dim3(256), 0, stream>>>(s, d, n);
  };
  cvt(Wq,  wq_bf,  (size_t)ND * ND);
  cvt(Wk,  wk_bf,  (size_t)ND * ND);
  cvt(Wv,  wv_bf,  (size_t)ND * ND);
  cvt(Wz,  wz_bf,  (size_t)ND * ND);
  cvt(Wo,  wo_bf,  (size_t)ND * ND);
  cvt(gdw, gdw_bf, (size_t)NR * ND);
  cvt(guw, guw_bf, (size_t)ND * NR);
  cvt(f1w, f1w_bf, (size_t)NFF * ND);
  cvt(f2w, f2w_bf, (size_t)ND * NFF);

  auto gemm = [&](const __hip_bfloat16* A, const __hip_bfloat16* W_,
                  const float* bias, float* oF, __hip_bfloat16* oB,
                  const float* res, int M, int N, int K, int op, float scale) {
    gemm_bf16_kernel<<<dim3(N / 128, M / 64), dim3(256), 0, stream>>>(
        A, W_, bias, oF, oB, res, M, N, K, op, scale);
  };

  // 1) norm1
  rmsnorm_bf16_kernel<<<dim3(NM), dim3(256), 0, stream>>>(x, n1w, xn_bf);
  // 2) projections (Q pre-scaled by 1/sqrt(DH)=1/8 for attention)
  gemm(xn_bf, wq_bf, bq, nullptr, q_bf, nullptr, NM, ND, ND, 0, 0.125f);
  gemm(xn_bf, wk_bf, bk, nullptr, k_bf, nullptr, NM, ND, ND, 0, 1.0f);
  gemm(xn_bf, wv_bf, bv, nullptr, v_bf, nullptr, NM, ND, ND, 0, 1.0f);
  gemm(xn_bf, wz_bf, bz, z_f,     z_bf, nullptr, NM, ND, ND, 0, 1.0f);
  // 3) decay gating MLP: relu(z@gdw^T+gdb) @ guw^T + gub -> sigmoid+clamp
  gemm(z_bf,  gdw_bf, gdb, nullptr, g1_bf, nullptr, NM, NR, ND, 1, 1.0f);
  gemm(g1_bf, guw_bf, gub, gamma,   nullptr, nullptr, NM, ND, NR, 3, 1.0f);
  // 4) V transpose for attention PV operand
  transpose_v_kernel<<<dim3(2048), dim3(256), 0, stream>>>(v_bf, vt_bf);
  // 5) sequential recurrence (consumes z_f, gamma)
  recurrence_kernel<<<dim3(NB), dim3(1024), 0, stream>>>(z_f, gamma, Wg, bg, hseq);
  // 6) attention (consumes q/k/vt; writes into z_bf region)
  attention_kernel<<<dim3(NS / 16, NH, NB), dim3(256), 0, stream>>>(q_bf, k_bf, vt_bf,
                                                                    aob_bf);
  // 7) output projection (writes into z_f region)
  gemm(aob_bf, wo_bf, bo, ao_f, nullptr, nullptr, NM, ND, ND, 0, 1.0f);
  // 8) residual y = x + ao + hseq (writes into gamma region)
  add3_kernel<<<dim3(2048), dim3(256), 0, stream>>>(x, ao_f, hseq, y,
                                                    (size_t)NM * ND);
  // 9) norm2 + FFN + final residual (ff1 reuses q..vt region; out = y + ff2)
  rmsnorm_bf16_kernel<<<dim3(NM), dim3(256), 0, stream>>>(y, n2w, yn_bf);
  gemm(yn_bf,  f1w_bf, f1b, nullptr, ff1_bf, nullptr, NM, NFF, ND, 2, 1.0f);
  gemm(ff1_bf, f2w_bf, f2b, out,     nullptr, y,      NM, ND, NFF, 0, 1.0f);
}
